// Block_21380347200224
// MI455X (gfx1250) — matrix operand order repaired and checked by reading, not compiled, not run on hardware
//
#include <hip/hip_runtime.h>
#include <hip/hip_bf16.h>
#include <math.h>

// ---------------- problem constants ----------------
#define BATCH 64
#define SH 9
#define SW 11
#define SD 9
#define SN 891            // 9*11*9
#define SC 900
#define SM (BATCH * SN)   // 57024 rows
#define MP 57088          // SM padded to 128
#define RANK 450
#define HID 1800
#define DR 5              // rfft last-dim bins

// padded sizes: K strides multiple of 32, N coverage multiple of 128 (block tile N)
#define CP 928            // K pad of 900
#define NP_R 512          // 450 -> 512   (t1 stride / u1,u2 weight rows ... )
#define NP_H 1920         // 1800 -> 1920 (t2 stride / v1 rows / u2 K)
#define NP_C 1024         // 900 -> 1024  (final grid coverage / v2 rows)

typedef __attribute__((ext_vector_type(16))) __bf16 v16bf;
typedef __attribute__((ext_vector_type(8)))  __bf16 v8bf;
typedef __attribute__((ext_vector_type(8)))  float  v8f;

// ---------------- LayerNorm over C=900 ----------------
template <bool OUTBF>
__global__ __launch_bounds__(256) void ln_kernel(const float* __restrict__ in,
                                                 const float* __restrict__ gamma,
                                                 const float* __restrict__ beta,
                                                 void* __restrict__ outp) {
  const int row = blockIdx.x;
  const float* p = in + (size_t)row * SC;
  int tid = threadIdx.x;
  float v[4];
  float s = 0.f, s2 = 0.f;
  #pragma unroll
  for (int i = 0; i < 4; ++i) {
    int c = tid + i * 256;
    float x = (c < SC) ? p[c] : 0.f;
    v[i] = x; s += x; s2 += x * x;
  }
  __shared__ float rs[256], rq[256];
  rs[tid] = s; rq[tid] = s2;
  __syncthreads();
  for (int o = 128; o > 0; o >>= 1) {
    if (tid < o) { rs[tid] += rs[tid + o]; rq[tid] += rq[tid + o]; }
    __syncthreads();
  }
  float mean = rs[0] * (1.f / SC);
  float var  = rq[0] * (1.f / SC) - mean * mean;
  float rstd = rsqrtf(var + 1e-5f);
  #pragma unroll
  for (int i = 0; i < 4; ++i) {
    int c = tid + i * 256;
    if (OUTBF) {
      if (c < CP) {
        float y = (c < SC) ? (v[i] - mean) * rstd * gamma[c] + beta[c] : 0.f;
        ((__bf16*)outp)[(size_t)row * CP + c] = (__bf16)y;
      }
    } else {
      if (c < SC)
        ((float*)outp)[(size_t)row * SC + c] = (v[i] - mean) * rstd * gamma[c] + beta[c];
    }
  }
}

// ---------------- 3D spectral filter ----------------
#define CC 20
#define NCHUNK 45
#define PI2 6.28318530717958647692f

__device__ inline void dft_line(float* re, float* im, int base, int stride, int L,
                                const float* mr, const float* mi, float csign) {
  float vr[11], vi[11];
  for (int i = 0; i < L; ++i) { vr[i] = re[base + i * stride]; vi[i] = im[base + i * stride]; }
  for (int k = 0; k < L; ++k) {
    float ar = 0.f, ai = 0.f;
    for (int n = 0; n < L; ++n) {
      float cr = mr[k * L + n];
      float ci = csign * mi[k * L + n];
      ar += vr[n] * cr - vi[n] * ci;
      ai += vr[n] * ci + vi[n] * cr;
    }
    re[base + k * stride] = ar;
    im[base + k * stride] = ai;
  }
}

extern __shared__ float fsmem[];

__global__ __launch_bounds__(256) void filter_kernel(const float* __restrict__ yn,
                                                     const float* __restrict__ cw,
                                                     float* __restrict__ yf) {
  const int b  = blockIdx.x / NCHUNK;
  const int c0 = (blockIdx.x % NCHUNK) * CC;
  const int tid = threadIdx.x;

  float* re  = fsmem;
  float* im  = re + SN * CC;
  float* m9r = im + SN * CC;
  float* m9i = m9r + 81;
  float* mbr = m9i + 81;
  float* mbi = mbr + 121;

  if (tid < 81)  { int k = tid / 9,  n = tid % 9;  float a = -PI2 * (float)(k * n) / 9.f;
                   m9r[tid] = cosf(a); m9i[tid] = sinf(a); }
  if (tid >= 128 && tid < 128 + 121) { int j = tid - 128; int k = j / 11, n = j % 11;
                   float a = -PI2 * (float)(k * n) / 11.f;
                   mbr[j] = cosf(a); mbi[j] = sinf(a); }
  __syncthreads();

  for (int l = tid; l < SW * SD * CC; l += 256) {
    int cc = l % CC; int t = l / CC; int d = t % SD; int w = t / SD;
    int c = c0 + cc;
    float v[SH];
    #pragma unroll
    for (int h = 0; h < SH; ++h)
      v[h] = yn[((size_t)b * SN + h * (SW * SD) + w * SD + d) * SC + c];
    #pragma unroll
    for (int k = 0; k < SH; ++k) {
      float ar = 0.f, ai = 0.f;
      #pragma unroll
      for (int h = 0; h < SH; ++h) { ar += v[h] * m9r[k * SH + h]; ai += v[h] * m9i[k * SH + h]; }
      int idx = ((k * SW + w) * SD + d) * CC + cc;
      re[idx] = ar; im[idx] = ai;
    }
  }
  __syncthreads();

  for (int l = tid; l < SH * SD * CC; l += 256) {
    int cc = l % CC; int t = l / CC; int d = t % SD; int h = t / SD;
    dft_line(re, im, (h * SW * SD + d) * CC + cc, SD * CC, SW, mbr, mbi, 1.f);
  }
  __syncthreads();
  for (int l = tid; l < SH * SW * CC; l += 256) {
    int cc = l % CC; int t = l / CC; int w = t % SW; int h = t / SW;
    dft_line(re, im, ((h * SW + w) * SD) * CC + cc, CC, SD, m9r, m9i, 1.f);
  }
  __syncthreads();

  for (int e = tid; e < SN * CC; e += 256) {
    int cc = e % CC; int t = e / CC; int d = t % SD; int w = (t / SD) % SW; int h = t / (SD * SW);
    int c = c0 + cc;
    int hh, ww, dr; float sgn;
    if (d < DR) { hh = h; ww = w; dr = d; sgn = 1.f; }
    else        { hh = (SH - h) % SH; ww = (SW - w) % SW; dr = SD - d; sgn = -1.f; }
    size_t wi_ = ((((size_t)hh * SW + ww) * DR + dr) * SC + c) * 2;
    float wr = cw[wi_], wimag = sgn * cw[wi_ + 1];
    float xr = re[e], xi = im[e];
    const float scl = 1.f / (float)SN;
    re[e] = (xr * wr - xi * wimag) * scl;
    im[e] = (xr * wimag + xi * wr) * scl;
  }
  __syncthreads();

  for (int l = tid; l < SH * SW * CC; l += 256) {
    int cc = l % CC; int t = l / CC; int w = t % SW; int h = t / SW;
    dft_line(re, im, ((h * SW + w) * SD) * CC + cc, CC, SD, m9r, m9i, -1.f);
  }
  __syncthreads();
  for (int l = tid; l < SH * SD * CC; l += 256) {
    int cc = l % CC; int t = l / CC; int d = t % SD; int h = t / SD;
    dft_line(re, im, (h * SW * SD + d) * CC + cc, SD * CC, SW, mbr, mbi, -1.f);
  }
  __syncthreads();
  for (int l = tid; l < SW * SD * CC; l += 256) {
    int cc = l % CC; int t = l / CC; int d = t % SD; int w = t / SD;
    dft_line(re, im, (w * SD + d) * CC + cc, SW * SD * CC, SH, m9r, m9i, -1.f);
  }
  __syncthreads();

  for (int e = tid; e < SN * CC; e += 256) {
    int cc = e % CC; int n = e / CC;
    yf[((size_t)b * SN + n) * SC + c0 + cc] = re[e];
  }
}

// ---------------- weight / bias padding kernels ----------------
__global__ void cvt_pad_bf16(const float* __restrict__ s, __bf16* __restrict__ d,
                             int N, int K, int Np, int Kp) {
  int i = blockIdx.x * 256 + threadIdx.x;
  if (i >= Np * Kp) return;
  int n = i / Kp, k = i % Kp;
  float v = (n < N && k < K) ? s[(size_t)n * K + k] : 0.f;
  d[i] = (__bf16)v;
}

__global__ void pad_bias_f32(const float* __restrict__ s, float* __restrict__ d, int N, int Np) {
  int i = blockIdx.x * 256 + threadIdx.x;
  if (i < Np) d[i] = (i < N) ? s[i] : 0.f;
}

// ---------------- bf16 WMMA GEMM ----------------
// Block tile 128x128, 8 waves, wave tile 32x64 (2x4 WMMA subtiles), BK=32,
// double-buffered LDS, bank-conflict-free padded stride (40 bf16 = 80B).
// GUARD=false: grid exactly covers [MP x ldOut]; no epilogue branches.
template <bool BIAS, bool GELU, bool RESID, bool OUTBF, bool GUARD>
__global__ __launch_bounds__(256) void gemm_bf16_wmma(const __bf16* __restrict__ A,
                                                      const __bf16* __restrict__ W,
                                                      const float* __restrict__ bias,
                                                      const float* __restrict__ resid,
                                                      void* __restrict__ outp,
                                                      int ldK, int Nreal, int Mreal,
                                                      int ldOut) {
  constexpr int BM = 128, BN = 128, BK = 32, BKP = 40;
  __shared__ __align__(16) __bf16 As[2][BM * BKP];
  __shared__ __align__(16) __bf16 Bs[2][BN * BKP];
  const int tid = threadIdx.x;
  const int lane = tid & 31;
  const int wave = tid >> 5;
  const int wm = wave & 3;   // 4 waves along M (32 rows each)
  const int wn = wave >> 2;  // 2 waves along N (64 cols each)
  const int m0 = blockIdx.y * BM;
  const int n0 = blockIdx.x * BN;
  const int nk = ldK / BK;

  // staging: A tile 128x32 = 512 v8bf chunks (2/thread), B tile 128x32 same
  const int r0  = tid >> 2;            // 0..63
  const int kc  = (tid & 3) * 8;       // 0,8,16,24
  const __bf16* arow0 = A + (size_t)(m0 + r0) * ldK + kc;
  const __bf16* arow1 = A + (size_t)(m0 + r0 + 64) * ldK + kc;
  const __bf16* brow0 = W + (size_t)(n0 + r0) * ldK + kc;
  const __bf16* brow1 = W + (size_t)(n0 + r0 + 64) * ldK + kc;
  const int sA0 = r0 * BKP + kc, sA1 = (r0 + 64) * BKP + kc;

  v8bf a0 = *(const v8bf*)arow0;
  v8bf a1 = *(const v8bf*)arow1;
  v8bf b0 = *(const v8bf*)brow0;
  v8bf b1 = *(const v8bf*)brow1;
  *(v8bf*)&As[0][sA0] = a0;
  *(v8bf*)&As[0][sA1] = a1;
  *(v8bf*)&Bs[0][sA0] = b0;
  *(v8bf*)&Bs[0][sA1] = b1;
  __syncthreads();

  v8f acc[2][4] = {};
  const int mrow = lane & 15;
  const int akb = (lane & 16) ? 8 : 0;   // A K-base per half-wave
  const int bkb = (lane & 16) ? 8 : 0;   // B K-base per half-wave

  for (int kt = 0; kt < nk; ++kt) {
    const int cur = kt & 1, nxt = cur ^ 1;
    const bool more = (kt + 1 < nk);
    if (more) {
      const int k0 = (kt + 1) * BK;
      a0 = *(const v8bf*)(arow0 + k0);
      a1 = *(const v8bf*)(arow1 + k0);
      b0 = *(const v8bf*)(brow0 + k0);
      b1 = *(const v8bf*)(brow1 + k0);
    }
    v16bf af[2], bfr[4];
    #pragma unroll
    for (int s = 0; s < 2; ++s) {
      const __bf16* p = &As[cur][(wm * 32 + s * 16 + mrow) * BKP + akb];
      v8bf lo = *(const v8bf*)p;
      v8bf hi = *(const v8bf*)(p + 16);
      af[s] = __builtin_shufflevector(lo, hi, 0,1,2,3,4,5,6,7,8,9,10,11,12,13,14,15);
    }
    #pragma unroll
    for (int t = 0; t < 4; ++t) {
      const __bf16* p = &Bs[cur][(wn * 64 + t * 16 + mrow) * BKP + bkb];
      v8bf lo = *(const v8bf*)p;
      v8bf hi = *(const v8bf*)(p + 16);
      bfr[t] = __builtin_shufflevector(lo, hi, 0,1,2,3,4,5,6,7,8,9,10,11,12,13,14,15);
    }
    #pragma unroll
    for (int s = 0; s < 2; ++s)
      #pragma unroll
      for (int t = 0; t < 4; ++t)
        acc[s][t] = __builtin_amdgcn_wmma_f32_16x16x32_bf16(
            false, af[s], false, bfr[t], (short)0, acc[s][t], false, false);
    if (more) {
      *(v8bf*)&As[nxt][sA0] = a0;
      *(v8bf*)&As[nxt][sA1] = a1;
      *(v8bf*)&Bs[nxt][sA0] = b0;
      *(v8bf*)&Bs[nxt][sA1] = b1;
    }
    __syncthreads();
  }

  // epilogue: C/D layout -> VGPR e: M = e (+8 upper half-wave), N = lane&15
  const int mbase = m0 + wm * 32 + ((lane & 16) ? 8 : 0);
  const int nbase = n0 + wn * 64 + mrow;
  #pragma unroll
  for (int s = 0; s < 2; ++s) {
    #pragma unroll
    for (int t = 0; t < 4; ++t) {
      #pragma unroll
      for (int e = 0; e < 8; ++e) {
        int gm = mbase + s * 16 + e;
        int gn = nbase + t * 16;
        float v = acc[s][t][e];
        if (BIAS) v += bias[gn];  // bias pre-padded with zeros
        if (GELU) v = 0.5f * v * (1.f + erff(v * 0.70710678118654752f));
        if (GUARD) {
          if (gm < Mreal && gn < Nreal) {
            if (RESID) v += resid[(size_t)gm * Nreal + gn];
            ((float*)outp)[(size_t)gm * ldOut + gn] = v;
          }
        } else {
          if (OUTBF) ((__bf16*)outp)[(size_t)gm * ldOut + gn] = (__bf16)v;
          else       ((float*)outp)[(size_t)gm * ldOut + gn] = v;
        }
      }
    }
  }
}

// ---------------- host-side orchestration ----------------
extern "C" void kernel_launch(void* const* d_in, const int* in_sizes, int n_in,
                              void* d_out, int out_size, void* d_ws, size_t ws_size,
                              hipStream_t stream) {
  (void)in_sizes; (void)n_in; (void)out_size; (void)ws_size;
  const float* x      = (const float*)d_in[0];
  const float* cw     = (const float*)d_in[1];
  const float* gamma1 = (const float*)d_in[2];
  const float* beta1  = (const float*)d_in[3];
  const float* gamma2 = (const float*)d_in[4];
  const float* beta2  = (const float*)d_in[5];
  const float* u1_w   = (const float*)d_in[6];
  const float* v1_w   = (const float*)d_in[7];
  const float* v1_b   = (const float*)d_in[8];
  const float* u2_w   = (const float*)d_in[9];
  const float* v2_w   = (const float*)d_in[10];
  const float* v2_b   = (const float*)d_in[11];
  float* out = (float*)d_out;

  // ---- workspace carve-up ----
  float* yn  = (float*)d_ws;                        // SM*900 fp32
  float* yf  = yn + (size_t)SM * SC;                // SM*900 fp32
  float* b1p = yf + (size_t)SM * SC;                // 1920 fp32 (padded v1_b)
  float* b2p = b1p + NP_H;                          // 1024 fp32 (padded v2_b)
  __bf16* zn  = (__bf16*)(b2p + NP_C);              // MP x CP
  __bf16* t1  = zn + (size_t)MP * CP;               // MP x NP_R
  __bf16* t2  = t1 + (size_t)MP * NP_R;             // MP x NP_H
  __bf16* u1b = t2 + (size_t)MP * NP_H;             // NP_R x CP
  __bf16* v1b = u1b + (size_t)NP_R * CP;            // NP_H x NP_R
  __bf16* u2b = v1b + (size_t)NP_H * NP_R;          // NP_R x NP_H
  __bf16* v2b = u2b + (size_t)NP_R * NP_H;          // NP_C x NP_R

  // 1) LayerNorm1 (fp32 out for spectral filter)
  ln_kernel<false><<<SM, 256, 0, stream>>>(x, gamma1, beta1, yn);

  // 2) spectral filter: 891x20 complex slab resident in ~141KB LDS per block
  const int SMEM = (2 * SN * CC + 81 * 2 + 121 * 2) * sizeof(float);
  filter_kernel<<<BATCH * NCHUNK, 256, SMEM, stream>>>(yn, cw, yf);

  // 3) LayerNorm2 fused with bf16 conversion + K-pad
  ln_kernel<true><<<SM, 256, 0, stream>>>(yf, gamma2, beta2, zn);

  // 4) weights -> bf16 padded; biases -> zero-padded fp32
  cvt_pad_bf16<<<(NP_R * CP + 255) / 256, 256, 0, stream>>>(u1_w, u1b, RANK, SC,   NP_R, CP);
  cvt_pad_bf16<<<(NP_H * NP_R + 255) / 256, 256, 0, stream>>>(v1_w, v1b, HID,  RANK, NP_H, NP_R);
  cvt_pad_bf16<<<(NP_R * NP_H + 255) / 256, 256, 0, stream>>>(u2_w, u2b, RANK, HID,  NP_R, NP_H);
  cvt_pad_bf16<<<(NP_C * NP_R + 255) / 256, 256, 0, stream>>>(v2_w, v2b, SC,   RANK, NP_C, NP_R);
  pad_bias_f32<<<(NP_H + 255) / 256, 256, 0, stream>>>(v1_b, b1p, HID, NP_H);
  pad_bias_f32<<<(NP_C + 255) / 256, 256, 0, stream>>>(v2_b, b2p, SC, NP_C);

  // 5) MLP chain: all K-loops and intermediate epilogues branch-free
  const int GM = MP / 128;  // 446
  dim3 blk(256);
  // t1 = zn * u1^T                 (K=928 -> 29 tiles; out 512-wide bf16)
  gemm_bf16_wmma<false, false, false, true, false><<<dim3(NP_R / 128, GM), blk, 0, stream>>>(
      zn, u1b, nullptr, nullptr, t1, CP, RANK, MP, NP_R);
  // t2 = gelu(t1 * v1^T + b1)      (K=512 -> 16 tiles; out 1920-wide bf16)
  gemm_bf16_wmma<true, true, false, true, false><<<dim3(NP_H / 128, GM), blk, 0, stream>>>(
      t1, v1b, b1p, nullptr, t2, NP_R, HID, MP, NP_H);
  // t1 = t2 * u2^T                 (K=1920 -> 60 tiles; overwrite t1)
  gemm_bf16_wmma<false, false, false, true, false><<<dim3(NP_R / 128, GM), blk, 0, stream>>>(
      t2, u2b, nullptr, nullptr, t1, NP_H, RANK, MP, NP_R);
  // out = t1 * v2^T + b2 + x       (K=512 -> 16 tiles; fp32 out, strict guards)
  gemm_bf16_wmma<true, false, true, false, true><<<dim3(NP_C / 128, GM), blk, 0, stream>>>(
      t1, v2b, b2p, x, out, NP_R, SC, SM, SC);
}
